// SwinTransformerBlock_68436008894541
// MI455X (gfx1250) — matrix operand order repaired and checked by reading, not compiled, not run on hardware
//
#include <hip/hip_runtime.h>
#include <math.h>

typedef __bf16 bf16_t;
typedef bf16_t v16bf __attribute__((ext_vector_type(16)));
typedef float  v8f   __attribute__((ext_vector_type(8)));

// ---------------- workspace layout (bytes) ----------------
#define OFF_QKVW  0                      // 576*192 bf16 = 221184
#define OFF_PROJW 221184                 // 192*192 bf16 = 73728
#define OFF_W1    294912                 // 768*192 bf16 = 294912
#define OFF_W2    589824                 // 192*768 bf16 = 294912
#define OFF_BIAS  884736                 // 6*49*49 f32  = 57624

// ---------------- WMMA operand loaders (16x16x32 bf16; this code's contiguous k order, A and B alike) ----------------
// A matrix 16x32 (MxK), 16-bit: lanes 0-15 -> M=lane, K = k0+0..15;
// lanes 16-31 -> M=lane-16, K = k0+16..31
__device__ __forceinline__ v16bf loadA16(const bf16_t* base, int stride, int m0, int k0,
                                         int lane, int mMax) {
  const int hh = lane >> 4;
  const int row = m0 + (lane & 15);
  v16bf a;
  if (row < mMax) {
    const bf16_t* p = base + row * stride + k0 + hh * 16;
#pragma unroll
    for (int e = 0; e < 16; ++e) a[e] = p[e];
  } else {
#pragma unroll
    for (int e = 0; e < 16; ++e) a[e] = (bf16_t)0.0f;
  }
  return a;
}

// B matrix 32x16 (KxN) from row-major W[N][K] (B[k][n] = W[n][k]):
// lanes 0-15 hold K=k0+0..15, lanes 16-31 hold K=k0+16..31, N = lane%16.
// -> one contiguous 32B read per lane.
__device__ __forceinline__ v16bf loadB_NK(const bf16_t* base, int stride, int n0, int k0,
                                          int lane) {
  const bf16_t* p = base + (n0 + (lane & 15)) * stride + k0 + ((lane >> 4) << 4);
  v16bf b;
#pragma unroll
  for (int e = 0; e < 16; ++e) b[e] = p[e];
  return b;
}

// B matrix 32x16 (KxN) from row-major V[K][N] (B[k][n] = V[k][n]) -> column gather
__device__ __forceinline__ v16bf loadB_KN(const bf16_t* base, int stride, int k0, int n0,
                                          int lane) {
  const bf16_t* p = base + (k0 + ((lane >> 4) << 4)) * stride + n0 + (lane & 15);
  v16bf b;
#pragma unroll
  for (int e = 0; e < 16; ++e) b[e] = p[e * stride];
  return b;
}

__device__ __forceinline__ v8f wmma_bf16(v16bf a, v16bf b, v8f c) {
  return __builtin_amdgcn_wmma_f32_16x16x32_bf16(false, a, false, b, (short)0, c, false, false);
}

// ---------------- kernel 0: weight bf16 conversion + rel-pos bias gather ----------------
__global__ void swin_prep_kernel(const float* __restrict__ qkv_w, const float* __restrict__ proj_w,
                                 const float* __restrict__ w1, const float* __restrict__ w2,
                                 const float* __restrict__ relt, char* __restrict__ ws) {
  bf16_t* qw = (bf16_t*)(ws + OFF_QKVW);
  bf16_t* pw = (bf16_t*)(ws + OFF_PROJW);
  bf16_t* m1 = (bf16_t*)(ws + OFF_W1);
  bf16_t* m2 = (bf16_t*)(ws + OFF_W2);
  float*  bg = (float*)(ws + OFF_BIAS);
  const int n0 = 576 * 192, n1 = 192 * 192, n2 = 768 * 192, n3 = 192 * 768, nb = 6 * 49 * 49;
  int i = blockIdx.x * 256 + threadIdx.x;
  if (i < n0)                     qw[i] = (bf16_t)qkv_w[i];
  else if (i < n0 + n1)           pw[i - n0] = (bf16_t)proj_w[i - n0];
  else if (i < n0 + n1 + n2)      m1[i - n0 - n1] = (bf16_t)w1[i - n0 - n1];
  else if (i < n0 + n1 + n2 + n3) m2[i - n0 - n1 - n2] = (bf16_t)w2[i - n0 - n1 - n2];
  else if (i < n0 + n1 + n2 + n3 + nb) {
    int t = i - (n0 + n1 + n2 + n3);
    int h = t / 2401, nm = t % 2401, n = nm / 49, m = nm % 49;
    int rel = ((n / 7) - (m / 7) + 6) * 13 + ((n % 7) - (m % 7) + 6);
    bg[(h * 49 + n) * 49 + m] = relt[rel * 6 + h];
  }
}

// ---------------- kernel 1: windowed attention (one window / workgroup) ----------------
// LDS: xb[64][200]bf16 | qkvb[64][584]bf16 | scores[6][49][52]f32 | P[6][49][72]bf16 | ao[64][200]bf16
#define K1_SMEM 229440
__global__ __launch_bounds__(256, 1)
void swin_attn_kernel(const float* __restrict__ x,
                      const bf16_t* __restrict__ qkvw, const float* __restrict__ qkv_b,
                      const bf16_t* __restrict__ projw, const float* __restrict__ proj_b,
                      const float* __restrict__ biasg, float* __restrict__ stage) {
  extern __shared__ char smem[];
  bf16_t* xb   = (bf16_t*)(smem + 0);        // stride 200
  bf16_t* qkvb = (bf16_t*)(smem + 25600);    // stride 584
  float*  sc   = (float*)(smem + 100352);    // stride 52
  bf16_t* Pm   = (bf16_t*)(smem + 161504);   // stride 72
  bf16_t* ao   = (bf16_t*)(smem + 203840);   // stride 200

  const int tid = threadIdx.x, lane = tid & 31, wave = tid >> 5, hh = lane >> 4;
  const int wid = blockIdx.x;
  const int b = wid >> 8, wy = (wid >> 4) & 15, wx = wid & 15;
  const int h0 = wy * 7, w0 = wx * 7;

  // zero padded token rows 49..63
  for (int idx = tid; idx < 15 * 200; idx += 256)
    xb[(49 + idx / 200) * 200 + (idx % 200)] = (bf16_t)0.0f;
  // gather window [49 x 192] from NCHW x -> bf16 LDS
  for (int idx = tid; idx < 192 * 49; idx += 256) {
    int c = idx / 49, n = idx % 49;
    float v = x[(((b * 192 + c) * 112) + (h0 + n / 7)) * 112 + (w0 + n % 7)];
    xb[n * 200 + c] = (bf16_t)v;
  }
  __syncthreads();

  // Phase A: qkv[64,576] = xb @ qkv_w^T + qkv_b
  // wave-strided tiles: tm = wave&3 is wave-invariant -> hoist all 6 A fragments.
  {
    const int tm = wave & 3;
    v16bf aT[6];
#pragma unroll
    for (int ks = 0; ks < 6; ++ks) aT[ks] = loadA16(xb, 200, tm * 16, ks * 32, lane, 64);
    for (int tn = (wave >> 2); tn < 36; tn += 2) {
      __builtin_prefetch(qkvw + (tn + 2 < 36 ? tn + 2 : tn) * 16 * 192 +
                         ((lane & 15) * 192) + ((lane >> 4) << 4), 0, 0);
      v8f acc = {};
#pragma unroll
      for (int ks = 0; ks < 6; ++ks)
        acc = wmma_bf16(aT[ks], loadB_NK(qkvw, 192, tn * 16, ks * 32, lane), acc);
      int col = tn * 16 + (lane & 15);
      float bias = qkv_b[col];
#pragma unroll
      for (int r = 0; r < 8; ++r)
        qkvb[(tm * 16 + r + hh * 8) * 584 + col] = (bf16_t)(acc[r] + bias);
    }
  }
  __syncthreads();

  // Phase B: one head per wave (waves 0..5)
  if (wave < 6) {
    const int h = wave;
    // scores S = q @ k^T * scale + bias  (q cols 0..191, k cols 192..383)
    // B tiles depend only on tn -> hoist the 4 of them over the tm loop.
    {
      v16bf bS[4];
#pragma unroll
      for (int tn = 0; tn < 4; ++tn)
        bS[tn] = loadB_NK(qkvb, 584, tn * 16, 192 + h * 32, lane);
      for (int tm = 0; tm < 4; ++tm) {
        v16bf a = loadA16(qkvb, 584, tm * 16, h * 32, lane, 64);
#pragma unroll
        for (int tn = 0; tn < 4; ++tn) {
          v8f acc = {};
          acc = wmma_bf16(a, bS[tn], acc);
          int m = tn * 16 + (lane & 15);
#pragma unroll
          for (int r = 0; r < 8; ++r) {
            int n = tm * 16 + r + hh * 8;
            if (n < 49 && m < 49)
              sc[(h * 49 + n) * 52 + m] = acc[r] * 0.1767766953f + biasg[(h * 49 + n) * 49 + m];
          }
        }
      }
    }
    // softmax rows (wave-private head; same-wave LDS ops stay ordered)
    for (int row = lane; row < 49; row += 32) {
      float* sr = sc + (h * 49 + row) * 52;
      float mx = -1e30f;
      for (int m = 0; m < 49; ++m) mx = fmaxf(mx, sr[m]);
      float sum = 0.0f;
      for (int m = 0; m < 49; ++m) { float e = __expf(sr[m] - mx); sr[m] = e; sum += e; }
      float rinv = 1.0f / sum;
      bf16_t* pr = Pm + (h * 49 + row) * 72;
      for (int m = 0; m < 49; ++m) pr[m] = (bf16_t)(sr[m] * rinv);
      for (int m = 49; m < 72; ++m) pr[m] = (bf16_t)0.0f;   // zero K-padding
    }
    // out = P @ v  (v cols 384..575); hoist the 4 strided V-column B tiles.
    {
      v16bf bV[2][2];
#pragma unroll
      for (int tn = 0; tn < 2; ++tn)
#pragma unroll
        for (int ks = 0; ks < 2; ++ks)
          bV[tn][ks] = loadB_KN(qkvb, 584, ks * 32, 384 + h * 32 + tn * 16, lane);
      for (int tm = 0; tm < 4; ++tm) {
        v16bf a0 = loadA16(Pm + h * 49 * 72, 72, tm * 16, 0, lane, 49);
        v16bf a1 = loadA16(Pm + h * 49 * 72, 72, tm * 16, 32, lane, 49);
#pragma unroll
        for (int tn = 0; tn < 2; ++tn) {
          v8f acc = {};
          acc = wmma_bf16(a0, bV[tn][0], acc);
          acc = wmma_bf16(a1, bV[tn][1], acc);
          int c = h * 32 + tn * 16 + (lane & 15);
#pragma unroll
          for (int r = 0; r < 8; ++r)
            ao[(tm * 16 + r + hh * 8) * 200 + c] = (bf16_t)acc[r];
        }
      }
    }
  }
  __syncthreads();

  // Phase C: proj + scatter to NCHW-flat staging (== d_out, overwritten by MLP kernel)
  {
    const int tm = wave & 3;
    v16bf aT[6];
#pragma unroll
    for (int ks = 0; ks < 6; ++ks) aT[ks] = loadA16(ao, 200, tm * 16, ks * 32, lane, 64);
    for (int tn = (wave >> 2); tn < 12; tn += 2) {
      v8f acc = {};
#pragma unroll
      for (int ks = 0; ks < 6; ++ks)
        acc = wmma_bf16(aT[ks], loadB_NK(projw, 192, tn * 16, ks * 32, lane), acc);
      int c = tn * 16 + (lane & 15);
      float pb = proj_b[c];
#pragma unroll
      for (int r = 0; r < 8; ++r) {
        int n = tm * 16 + r + hh * 8;
        if (n < 49)
          stage[(((b * 192 + c) * 112) + (h0 + n / 7)) * 112 + (w0 + n % 7)] = acc[r] + pb;
      }
    }
  }
}

// ---------------- kernel 2: LN1 -> MLP(GELU) -> LN2, in place on 64-row blocks ----------------
// LDS: xln[64][200]bf16 | {xf[64][192]f32 / hb[64][776]bf16 overlay} | y[64][192]f32 | reductions
#define K2_SMEM 176640
__global__ __launch_bounds__(256, 1)
void swin_mlp_kernel(float* __restrict__ io,
                     const bf16_t* __restrict__ w1bf, const float* __restrict__ b1,
                     const bf16_t* __restrict__ w2bf, const float* __restrict__ b2,
                     const float* __restrict__ n1w, const float* __restrict__ n1b,
                     const float* __restrict__ n2w, const float* __restrict__ n2b) {
  extern __shared__ char smem[];
  bf16_t* xln  = (bf16_t*)(smem + 0);        // stride 200
  float*  xf   = (float*)(smem + 25600);     // stride 192 (dead after LN1)
  bf16_t* hb   = (bf16_t*)(smem + 25600);    // stride 776 (overlays xf)
  float*  yb   = (float*)(smem + 124928);    // stride 192
  float*  redS = (float*)(smem + 174080);
  float*  redQ = (float*)(smem + 175104);
  float*  mu   = (float*)(smem + 176128);
  float*  rsd  = (float*)(smem + 176384);
  const int tid = threadIdx.x, lane = tid & 31, wave = tid >> 5, hh = lane >> 4;
  float* rows = io + (size_t)blockIdx.x * 64 * 192;

  for (int idx = tid; idx < 64 * 192; idx += 256) xf[idx] = rows[idx];
  __syncthreads();
  // LN1 statistics: 4 threads per row
  {
    const float* xr = xf + (tid >> 2) * 192 + (tid & 3) * 48;
    float s = 0.f, ss = 0.f;
    for (int i = 0; i < 48; ++i) { float v = xr[i]; s += v; ss += v * v; }
    redS[tid] = s; redQ[tid] = ss;
  }
  __syncthreads();
  if (tid < 64) {
    float S = 0.f, Q = 0.f;
    for (int j = 0; j < 4; ++j) { S += redS[tid * 4 + j]; Q += redQ[tid * 4 + j]; }
    float mean = S * (1.0f / 192.0f);
    mu[tid] = mean; rsd[tid] = rsqrtf(Q * (1.0f / 192.0f) - mean * mean + 1e-5f);
  }
  __syncthreads();
  for (int idx = tid; idx < 64 * 192; idx += 256) {
    int row = idx / 192, c = idx % 192;
    xln[row * 200 + c] = (bf16_t)((xf[idx] - mu[row]) * rsd[row] * n1w[c] + n1b[c]);
  }
  __syncthreads();
  // fc1 [64,192]x[192,768] + exact GELU; tm wave-invariant -> hoist 6 A fragments
  {
    const int tm = wave & 3;
    v16bf aT[6];
#pragma unroll
    for (int ks = 0; ks < 6; ++ks) aT[ks] = loadA16(xln, 200, tm * 16, ks * 32, lane, 64);
    for (int tn = (wave >> 2); tn < 48; tn += 2) {
      __builtin_prefetch(w1bf + (tn + 2 < 48 ? tn + 2 : tn) * 16 * 192 +
                         ((lane & 15) * 192) + ((lane >> 4) << 4), 0, 0);
      v8f acc = {};
#pragma unroll
      for (int ks = 0; ks < 6; ++ks)
        acc = wmma_bf16(aT[ks], loadB_NK(w1bf, 192, tn * 16, ks * 32, lane), acc);
      int c = tn * 16 + (lane & 15);
      float bias = b1[c];
#pragma unroll
      for (int r = 0; r < 8; ++r) {
        float v = acc[r] + bias;
        v = 0.5f * v * (1.0f + erff(v * 0.70710678118f));
        hb[(tm * 16 + r + hh * 8) * 776 + c] = (bf16_t)v;
      }
    }
  }
  __syncthreads();
  // fc2 [64,768]x[768,192]: K blocked 4x(6x32); 6 persistent accumulators per wave
  {
    const int tm = wave & 3;
    const int tnBase = wave >> 2;
    v8f acc[6];
#pragma unroll
    for (int t = 0; t < 6; ++t) acc[t] = (v8f){};
    for (int kb = 0; kb < 4; ++kb) {
      v16bf aT[6];
#pragma unroll
      for (int k2 = 0; k2 < 6; ++k2)
        aT[k2] = loadA16(hb, 776, tm * 16, (kb * 6 + k2) * 32, lane, 64);
#pragma unroll
      for (int t = 0; t < 6; ++t) {
        int tn = tnBase + 2 * t;
#pragma unroll
        for (int k2 = 0; k2 < 6; ++k2)
          acc[t] = wmma_bf16(aT[k2],
                             loadB_NK(w2bf, 768, tn * 16, (kb * 6 + k2) * 32, lane), acc[t]);
      }
    }
#pragma unroll
    for (int t = 0; t < 6; ++t) {
      int tn = tnBase + 2 * t;
      int c = tn * 16 + (lane & 15);
      float bias = b2[c];
#pragma unroll
      for (int r = 0; r < 8; ++r)
        yb[(tm * 16 + r + hh * 8) * 192 + c] = acc[t][r] + bias;
    }
  }
  __syncthreads();
  // LN2 + writeback
  {
    const float* xr = yb + (tid >> 2) * 192 + (tid & 3) * 48;
    float s = 0.f, ss = 0.f;
    for (int i = 0; i < 48; ++i) { float v = xr[i]; s += v; ss += v * v; }
    redS[tid] = s; redQ[tid] = ss;
  }
  __syncthreads();
  if (tid < 64) {
    float S = 0.f, Q = 0.f;
    for (int j = 0; j < 4; ++j) { S += redS[tid * 4 + j]; Q += redQ[tid * 4 + j]; }
    float mean = S * (1.0f / 192.0f);
    mu[tid] = mean; rsd[tid] = rsqrtf(Q * (1.0f / 192.0f) - mean * mean + 1e-5f);
  }
  __syncthreads();
  for (int idx = tid; idx < 64 * 192; idx += 256) {
    int row = idx / 192, c = idx % 192;
    rows[idx] = (yb[idx] - mu[row]) * rsd[row] * n2w[c] + n2b[c];
  }
}

// ---------------- host launch ----------------
extern "C" void kernel_launch(void* const* d_in, const int* in_sizes, int n_in,
                              void* d_out, int out_size, void* d_ws, size_t ws_size,
                              hipStream_t stream) {
  (void)in_sizes; (void)n_in; (void)out_size; (void)ws_size;
  const float* x      = (const float*)d_in[0];
  const float* qkv_w  = (const float*)d_in[1];
  const float* qkv_b  = (const float*)d_in[2];
  const float* proj_w = (const float*)d_in[3];
  const float* proj_b = (const float*)d_in[4];
  const float* relt   = (const float*)d_in[5];
  const float* n1w    = (const float*)d_in[6];
  const float* n1b    = (const float*)d_in[7];
  const float* n2w    = (const float*)d_in[8];
  const float* n2b    = (const float*)d_in[9];
  const float* w1     = (const float*)d_in[10];
  const float* b1     = (const float*)d_in[11];
  const float* w2     = (const float*)d_in[12];
  const float* b2     = (const float*)d_in[13];
  float* out = (float*)d_out;
  char*  ws  = (char*)d_ws;

  const bf16_t* qkvw_bf = (const bf16_t*)(ws + OFF_QKVW);
  const bf16_t* projw_bf= (const bf16_t*)(ws + OFF_PROJW);
  const bf16_t* w1_bf   = (const bf16_t*)(ws + OFF_W1);
  const bf16_t* w2_bf   = (const bf16_t*)(ws + OFF_W2);
  const float*  biasg   = (const float*)(ws + OFF_BIAS);

  hipFuncSetAttribute((const void*)swin_attn_kernel,
                      hipFuncAttributeMaxDynamicSharedMemorySize, K1_SMEM);
  hipFuncSetAttribute((const void*)swin_mlp_kernel,
                      hipFuncAttributeMaxDynamicSharedMemorySize, K2_SMEM);

  const int prepTotal = 576 * 192 + 192 * 192 + 768 * 192 + 192 * 768 + 6 * 49 * 49;
  swin_prep_kernel<<<(prepTotal + 255) / 256, 256, 0, stream>>>(qkv_w, proj_w, w1, w2, relt, ws);

  // 2048 windows: 8 batches x 16x16 windows; stages fp32 into d_out (NCHW-flat)
  swin_attn_kernel<<<2048, 256, K1_SMEM, stream>>>(x, qkvw_bf, qkv_b, projw_bf, proj_b,
                                                   biasg, out);
  // 100352 rows / 64 per WG = 1568; in-place on d_out
  swin_mlp_kernel<<<1568, 256, K2_SMEM, stream>>>(out, w1_bf, b1, w2_bf, b2,
                                                  n1w, n1b, n2w, n2b);
}
